// HyperbolicTransformerLayer_64622077935755
// MI455X (gfx1250) — matrix operand order repaired and checked by reading, not compiled, not run on hardware
//
#include <hip/hip_runtime.h>
#include <hip/hip_bf16.h>

#define NN 50000
#define EE 600000
#define DD 128
#define HH 8
#define DH 16
#define FF 512

typedef __attribute__((ext_vector_type(16))) _Float16 v16h;
typedef __attribute__((ext_vector_type(8)))  _Float16 v8h;
typedef __attribute__((ext_vector_type(8)))  float    v8f;

// ---------- helpers ----------
__device__ __forceinline__ unsigned fenc(float f) {
    unsigned u = __float_as_uint(f);
    return (u & 0x80000000u) ? ~u : (u | 0x80000000u);
}
__device__ __forceinline__ float fdec(unsigned u) {
    return (u & 0x80000000u) ? __uint_as_float(u & 0x7FFFFFFFu) : __uint_as_float(~u);
}

// ---------- conversion / transpose ----------
__global__ void cvt_f16_kernel(const float* __restrict__ in, _Float16* __restrict__ out, int n) {
    int i = blockIdx.x * blockDim.x + threadIdx.x;
    if (i < n) out[i] = (_Float16)in[i];
}

// W is K x Ncols (row-major, fp32). Wt is Ncols x K (row-major, f16): Wt[n*K+k] = W[k*Ncols+n]
__global__ void wtr_kernel(const float* __restrict__ W, _Float16* __restrict__ Wt, int K, int Ncols) {
    int i = blockIdx.x * blockDim.x + threadIdx.x;
    if (i >= K * Ncols) return;
    int k = i / Ncols, n = i % Ncols;
    Wt[(size_t)n * K + k] = (_Float16)W[i];
}

// ---------- WMMA GEMM: out[M x Ncols] = A[M x K] * Bt^T, Bt is Ncols x K ----------
// All epilogue behavior is compile-time: no scalar branches around stores.
template <int K, bool HAS_BIAS, bool RELU, bool OUT32, bool OUT16>
__global__ void gemm_wmma_kernel(const _Float16* __restrict__ A,
                                 const _Float16* __restrict__ Bt,
                                 const float* __restrict__ bias,
                                 float* __restrict__ out32,
                                 _Float16* __restrict__ out16,
                                 int M, int Ncols) {
    const int lane = threadIdx.x & 31;
    const int wave = threadIdx.x >> 5;
    const int tile_m = blockIdx.y * 8 + wave;     // wave-uniform guard: EXEC stays full
    if (tile_m * 16 >= M) return;
    const int tile_n = blockIdx.x;
    const int lh = lane & 15;
    const int hs = lane >> 4;

    // A fragment: lane holds row m = tile_m*16+lh; 16 contiguous K at k0 + hs*16
    const _Float16* __restrict__ Ap = A + (size_t)(tile_m * 16 + lh) * K + hs * 16;
    // B fragment: lane holds column n = tile_n*16+lh; 16 contiguous K at k0 + hs*16
    const _Float16* __restrict__ Bp = Bt + (size_t)(tile_n * 16 + lh) * K + hs * 16;

    v8f acc = {};
#pragma unroll
    for (int k0 = 0; k0 < K; k0 += 32) {
        v8h alo = *(const v8h*)(Ap + k0);
        v8h ahi = *(const v8h*)(Ap + k0 + 8);
        v16h b  = *(const v16h*)(Bp + k0);
        v16h a;
#pragma unroll
        for (int j = 0; j < 8; ++j) { a[j] = alo[j]; a[j + 8] = ahi[j]; }
        acc = __builtin_amdgcn_wmma_f32_16x16x32_f16(
            false, a, false, b, (short)0, acc, false, false);
    }

    const int n  = tile_n * 16 + lh;
    const int mb = tile_m * 16 + hs * 8;
    float bv = 0.0f;
    if (HAS_BIAS) bv = bias[n];
#pragma unroll
    for (int r = 0; r < 8; ++r) {
        float vv = acc[r] + bv;
        if (RELU) vv = fmaxf(vv, 0.0f);
        size_t idx = (size_t)(mb + r) * Ncols + n;
        if (OUT32) out32[idx] = vv;
        if (OUT16) out16[idx] = (_Float16)vv;
    }
}

// ---------- attention segment softmax ----------
__global__ void init_seg_kernel(unsigned* __restrict__ mEnc, float* __restrict__ denom) {
    int i = blockIdx.x * blockDim.x + threadIdx.x;
    if (i < NN * HH) { mEnc[i] = 0x007FFFFFu; /* enc(-inf) */ denom[i] = 0.0f; }
}
__global__ void zero_f32_kernel(float* __restrict__ p, int n) {
    int i = blockIdx.x * blockDim.x + threadIdx.x;
    if (i < n) p[i] = 0.0f;
}

__global__ void attn_scores_kernel(const float* __restrict__ q, const float* __restrict__ k,
                                   const int* __restrict__ ei,
                                   float* __restrict__ scores, unsigned* __restrict__ mEnc) {
    int idx = blockIdx.x * blockDim.x + threadIdx.x;
    if (idx >= EE * HH) return;
    int e = idx >> 3, h = idx & 7;
    int src = ei[e], dst = ei[EE + e];
    const float* qp = q + (size_t)dst * DD + h * DH;
    const float* kp = k + (size_t)src * DD + h * DH;
    float s = 0.0f;
#pragma unroll
    for (int d = 0; d < DH; ++d) s += qp[d] * kp[d];
    s *= 0.25f;                       // 1/sqrt(16)
    scores[idx] = s;
    atomicMax(mEnc + (size_t)dst * HH + h, fenc(s));
}

__global__ void attn_exp_kernel(float* __restrict__ scores, const int* __restrict__ ei,
                                const unsigned* __restrict__ mEnc, float* __restrict__ denom) {
    int idx = blockIdx.x * blockDim.x + threadIdx.x;
    if (idx >= EE * HH) return;
    int e = idx >> 3, h = idx & 7;
    int dst = ei[EE + e];
    float m  = fdec(mEnc[(size_t)dst * HH + h]);
    float ex = __expf(scores[idx] - m);
    scores[idx] = ex;
    atomicAdd(denom + (size_t)dst * HH + h, ex);
}

__global__ void attn_scatter_kernel(const float* __restrict__ scores, const float* __restrict__ denom,
                                    const float* __restrict__ v, const int* __restrict__ ei,
                                    float* __restrict__ attn) {
    int t = blockIdx.x * blockDim.x + threadIdx.x;
    if (t >= EE * DD) return;
    int e = t >> 7, d = t & 127, h = d >> 4;
    int src = ei[e], dst = ei[EE + e];
    float alpha = scores[(size_t)e * HH + h] / (denom[(size_t)dst * HH + h] + 1e-16f);
    atomicAdd(attn + (size_t)dst * DD + d, alpha * v[(size_t)src * DD + d]);
}

// ---------- fused uhg_add ((a+b)/2) + layernorm, wave per node ----------
template <bool OUT16>
__global__ void add_ln_kernel(const float* __restrict__ a, const float* __restrict__ b,
                              const float* __restrict__ g, const float* __restrict__ be,
                              float* __restrict__ out32, _Float16* __restrict__ out16) {
    int lane = threadIdx.x & 31, wave = threadIdx.x >> 5;
    int node = blockIdx.x * 8 + wave;
    if (node >= NN) return;
    const float4 av = *(const float4*)(a + (size_t)node * DD + lane * 4);
    const float4 bv = *(const float4*)(b + (size_t)node * DD + lane * 4);
    float z0 = (av.x + bv.x) * 0.5f, z1 = (av.y + bv.y) * 0.5f;
    float z2 = (av.z + bv.z) * 0.5f, z3 = (av.w + bv.w) * 0.5f;
    float s = z0 + z1 + z2 + z3;
#pragma unroll
    for (int m = 16; m > 0; m >>= 1) s += __shfl_xor(s, m, 32);
    float mean = s * (1.0f / 128.0f);
    float d0 = z0 - mean, d1 = z1 - mean, d2 = z2 - mean, d3 = z3 - mean;
    float vs = d0 * d0 + d1 * d1 + d2 * d2 + d3 * d3;
#pragma unroll
    for (int m = 16; m > 0; m >>= 1) vs += __shfl_xor(vs, m, 32);
    float rs = rsqrtf(vs * (1.0f / 128.0f) + 1e-5f);
    const float4 gv  = *(const float4*)(g  + lane * 4);
    const float4 bev = *(const float4*)(be + lane * 4);
    float o0 = gv.x * d0 * rs + bev.x, o1 = gv.y * d1 * rs + bev.y;
    float o2 = gv.z * d2 * rs + bev.z, o3 = gv.w * d3 * rs + bev.w;
    float4* op = (float4*)(out32 + (size_t)node * DD + lane * 4);
    *op = make_float4(o0, o1, o2, o3);
    if (OUT16) {
        _Float16* hp = out16 + (size_t)node * DD + lane * 4;
        hp[0] = (_Float16)o0; hp[1] = (_Float16)o1;
        hp[2] = (_Float16)o2; hp[3] = (_Float16)o3;
    }
}

// ---------- cross-ratio ----------
// homog==0: cr over raw 128-dim rows of x -> crbuf[0]
// homog==1: cr over rows of y with +1 homogeneous term; compute scale -> crbuf[1]
__global__ void cross_ratio_kernel(const float* __restrict__ x, float* __restrict__ crbuf, int homog) {
    __shared__ float sac[128], sbd[128], sad[128], sbc[128];
    int t = threadIdx.x;
    const float* x0 = x;
    const float* x1 = x + DD;
    const float* x2 = x + 2 * DD;
    const float* x3 = x + 3 * DD;
    sac[t] = x0[t] * x2[t]; sbd[t] = x1[t] * x3[t];
    sad[t] = x0[t] * x3[t]; sbc[t] = x1[t] * x2[t];
    __syncthreads();
    for (int s = 64; s > 0; s >>= 1) {
        if (t < s) { sac[t] += sac[t + s]; sbd[t] += sbd[t + s];
                     sad[t] += sad[t + s]; sbc[t] += sbc[t + s]; }
        __syncthreads();
    }
    if (t == 0) {
        float hc = homog ? 1.0f : 0.0f;
        float ac = sac[0] + hc, bd = sbd[0] + hc, ad = sad[0] + hc, bc = sbc[0] + hc;
        float cr = ac * bd / (ad * bc + 1e-8f);
        if (!homog) {
            crbuf[0] = cr;
        } else {
            float cr0 = crbuf[0];
            bool valid = __builtin_isfinite(cr) && __builtin_isfinite(cr0) && (fabsf(cr) > 1e-8f);
            float crs = valid ? cr : 1.0f;
            crbuf[1] = valid ? powf(fabsf(cr0 / crs), 0.25f) : 1.0f;
        }
    }
}

__global__ void scale_kernel(float* __restrict__ y, const float* __restrict__ crbuf, int n) {
    int i = blockIdx.x * blockDim.x + threadIdx.x;
    if (i < n) y[i] *= crbuf[1];
}

// ---------- launch ----------
extern "C" void kernel_launch(void* const* d_in, const int* in_sizes, int n_in,
                              void* d_out, int out_size, void* d_ws, size_t ws_size,
                              hipStream_t stream) {
    const float* x   = (const float*)d_in[0];
    const int*   ei  = (const int*)d_in[1];
    const float* Wq  = (const float*)d_in[2];
    const float* Wk  = (const float*)d_in[3];
    const float* Wv  = (const float*)d_in[4];
    const float* W1  = (const float*)d_in[5];
    const float* b1  = (const float*)d_in[6];
    const float* W2  = (const float*)d_in[7];
    const float* b2  = (const float*)d_in[8];
    const float* g1  = (const float*)d_in[9];
    const float* be1 = (const float*)d_in[10];
    const float* g2  = (const float*)d_in[11];
    const float* be2 = (const float*)d_in[12];
    float* out = (float*)d_out;

    char* w = (char*)d_ws;
    size_t off = 0;
    auto alloc = [&](size_t bytes) -> void* {
        off = (off + 255) & ~(size_t)255;
        void* p = w + off; off += bytes; return p;
    };
    float*    q    = (float*)alloc((size_t)NN * DD * 4);
    float*    kbuf = (float*)alloc((size_t)NN * DD * 4);
    float*    vbuf = (float*)alloc((size_t)NN * DD * 4);
    _Float16* x16  = (_Float16*)alloc((size_t)NN * DD * 2);
    _Float16* Wqt  = (_Float16*)alloc((size_t)DD * DD * 2);
    _Float16* Wkt  = (_Float16*)alloc((size_t)DD * DD * 2);
    _Float16* Wvt  = (_Float16*)alloc((size_t)DD * DD * 2);
    _Float16* W1t  = (_Float16*)alloc((size_t)FF * DD * 2);  // 512 x 128
    _Float16* W2t  = (_Float16*)alloc((size_t)DD * FF * 2);  // 128 x 512
    float*    scr  = (float*)alloc((size_t)EE * HH * 4);
    unsigned* mEnc = (unsigned*)alloc((size_t)NN * HH * 4);
    float*    den  = (float*)alloc((size_t)NN * HH * 4);
    float*    xln  = (float*)alloc((size_t)NN * DD * 4);
    _Float16* h16  = (_Float16*)alloc((size_t)NN * FF * 2);
    float*    crb  = (float*)alloc(256);
    // aliases (lifetimes disjoint): q dead after scores pass; k dead after scores pass;
    // x16 dead after QKV GEMM.
    float*    attn  = q;
    float*    ffout = kbuf;
    _Float16* xln16 = x16;

    const int TB = 256;
    dim3 blk(TB);

    // 1) conversions
    cvt_f16_kernel<<<dim3((NN * DD + TB - 1) / TB), blk, 0, stream>>>(x, x16, NN * DD);
    wtr_kernel<<<dim3((DD * DD + TB - 1) / TB), blk, 0, stream>>>(Wq, Wqt, DD, DD);
    wtr_kernel<<<dim3((DD * DD + TB - 1) / TB), blk, 0, stream>>>(Wk, Wkt, DD, DD);
    wtr_kernel<<<dim3((DD * DD + TB - 1) / TB), blk, 0, stream>>>(Wv, Wvt, DD, DD);
    wtr_kernel<<<dim3((DD * FF + TB - 1) / TB), blk, 0, stream>>>(W1, W1t, DD, FF);
    wtr_kernel<<<dim3((FF * DD + TB - 1) / TB), blk, 0, stream>>>(W2, W2t, FF, DD);

    // 2) cr0 from raw x
    cross_ratio_kernel<<<dim3(1), dim3(128), 0, stream>>>(x, crb, 0);

    // 3) Q/K/V GEMMs (WMMA f16 -> f32)
    dim3 gq(DD / 16, (NN / 16 + 7) / 8);
    gemm_wmma_kernel<DD, false, false, true, false>
        <<<gq, blk, 0, stream>>>(x16, Wqt, nullptr, q,    nullptr, NN, DD);
    gemm_wmma_kernel<DD, false, false, true, false>
        <<<gq, blk, 0, stream>>>(x16, Wkt, nullptr, kbuf, nullptr, NN, DD);
    gemm_wmma_kernel<DD, false, false, true, false>
        <<<gq, blk, 0, stream>>>(x16, Wvt, nullptr, vbuf, nullptr, NN, DD);

    // 4) segment softmax attention
    init_seg_kernel<<<dim3((NN * HH + TB - 1) / TB), blk, 0, stream>>>(mEnc, den);
    attn_scores_kernel<<<dim3((EE * HH + TB - 1) / TB), blk, 0, stream>>>(q, kbuf, ei, scr, mEnc);
    zero_f32_kernel<<<dim3((NN * DD + TB - 1) / TB), blk, 0, stream>>>(attn, NN * DD); // reuses q
    attn_exp_kernel<<<dim3((EE * HH + TB - 1) / TB), blk, 0, stream>>>(scr, ei, mEnc, den);
    attn_scatter_kernel<<<dim3((EE * DD + TB - 1) / TB), blk, 0, stream>>>(scr, den, vbuf, ei, attn);

    // 5) residual (uhg) + LN1 -> xln (f32) and xln16 (f16 for FFN GEMM)
    add_ln_kernel<true><<<dim3(NN / 8), blk, 0, stream>>>(x, attn, g1, be1, xln, xln16);

    // 6) FFN: GEMM1 (bias+relu, f16 out), GEMM2 (bias, f32 out)
    dim3 gf1(FF / 16, (NN / 16 + 7) / 8);
    gemm_wmma_kernel<DD, true, true, false, true>
        <<<gf1, blk, 0, stream>>>(xln16, W1t, b1, nullptr, h16, NN, FF);
    dim3 gf2(DD / 16, (NN / 16 + 7) / 8);
    gemm_wmma_kernel<FF, true, false, true, false>
        <<<gf2, blk, 0, stream>>>(h16, W2t, b2, ffout, nullptr, NN, DD);

    // 7) residual (uhg) + LN2 -> out
    add_ln_kernel<false><<<dim3(NN / 8), blk, 0, stream>>>(xln, ffout, g2, be2, out, nullptr);

    // 8) restore cross ratio: compute scale, then scale output
    cross_ratio_kernel<<<dim3(1), dim3(128), 0, stream>>>(out, crb, 1);
    scale_kernel<<<dim3((NN * DD + TB - 1) / TB), blk, 0, stream>>>(out, crb, NN * DD);
}
